// GPT_25950192402668
// MI455X (gfx1250) — compile-verified, matrix operand order checked by reading, not run on hardware
//
#include <hip/hip_runtime.h>
#include <hip/hip_bf16.h>
#include <stdint.h>

#define LNUM 4
#define BB 2
#define TT 2048
#define CC 512
#define HH 8
#define VV 32000
#define BT (BB*TT)
#define EPSF 1e-5f
#define NEG_INF (-3.0e38f)

typedef __attribute__((ext_vector_type(16))) __bf16 bf16x16;
typedef __attribute__((ext_vector_type(8)))  __bf16 bf16x8;
typedef __attribute__((ext_vector_type(8)))  float  f32x8;
typedef __attribute__((ext_vector_type(8)))  unsigned short u16x8;

__device__ __forceinline__ unsigned short f2bf(float f) {
  unsigned u = __float_as_uint(f);
  u += 0x7FFFu + ((u >> 16) & 1u);           // round-to-nearest-even
  return (unsigned short)(u >> 16);
}

// Load a 16-element bf16 WMMA fragment (two 16B chunks) -> v16bf
__device__ __forceinline__ bf16x16 ldfrag(const unsigned short* p0, const unsigned short* p1) {
  union U { u16x8 u; bf16x8 b; };
  U c0, c1;
  c0.u = *(const u16x8*)p0;
  c1.u = *(const u16x8*)p1;
  return __builtin_shufflevector(c0.b, c1.b, 0,1,2,3,4,5,6,7,8,9,10,11,12,13,14,15);
}

// Async 16B copy global -> LDS (CDNA5 GLOBAL_LOAD_ASYNC_TO_LDS_B128, ASYNCcnt)
__device__ __forceinline__ void async_cp16(unsigned lds_addr, const void* gaddr) {
  asm volatile("global_load_async_to_lds_b128 %0, %1, off"
               :: "v"(lds_addr), "v"(gaddr) : "memory");
}
__device__ __forceinline__ void wait_async_le4() {
  asm volatile("s_wait_asynccnt 0x4" ::: "memory");
}
__device__ __forceinline__ void wait_async_0() {
  asm volatile("s_wait_asynccnt 0x0" ::: "memory");
}

// ---------------------------------------------------------------------------
// Generic GEMM: C[M,N] = A[M,K](bf16) x Bw[N,K](bf16)^T, fp32 accumulate.
// mode 0: Cf = acc           (plain fp32 out)
// mode 1: Cbf = bf16(gelu(acc))
// mode 2: Cf = acc + resid   (residual add)
// Block tile 128x128, 8 waves each 32x64 (2x4 WMMA), K step 32.
// Double-buffered LDS filled by GLOBAL_LOAD_ASYNC_TO_LDS_B128.
// ---------------------------------------------------------------------------
__global__ __launch_bounds__(256) void k_gemm(
    const unsigned short* __restrict__ A, const unsigned short* __restrict__ Bw,
    float* __restrict__ Cf, unsigned short* __restrict__ Cbf,
    const float* __restrict__ resid,
    int M, int N, int K, int mode)
{
  __shared__ __align__(16) unsigned short lA[2][128 * 32];
  __shared__ __align__(16) unsigned short lB[2][128 * 32];
  const int tid  = threadIdx.x;
  const int lane = tid & 31, wave = tid >> 5;
  const int half = lane >> 4, lr = lane & 15;
  const int wm = wave >> 1, wn = wave & 1;     // 4 x 2 waves over 128x128
  const int bm = blockIdx.y, bn = blockIdx.x;

  // this thread's two 16B chunks per 128x32 tile
  const int c0row = tid >> 2,        c0off = (tid & 3) * 8;
  const int c1row = (tid + 256) >> 2, c1off = ((tid + 256) & 3) * 8;

  auto issue = [&](int k0, int buf) {
    async_cp16((unsigned)(uintptr_t)&lA[buf][c0row * 32 + c0off],
               &A[(size_t)(bm * 128 + c0row) * K + k0 + c0off]);
    async_cp16((unsigned)(uintptr_t)&lA[buf][c1row * 32 + c1off],
               &A[(size_t)(bm * 128 + c1row) * K + k0 + c1off]);
    async_cp16((unsigned)(uintptr_t)&lB[buf][c0row * 32 + c0off],
               &Bw[(size_t)(bn * 128 + c0row) * K + k0 + c0off]);
    async_cp16((unsigned)(uintptr_t)&lB[buf][c1row * 32 + c1off],
               &Bw[(size_t)(bn * 128 + c1row) * K + k0 + c1off]);
  };

  f32x8 acc[2][4] = {};
  issue(0, 0);
  int cur = 0;
  for (int k0 = 0; k0 < K; k0 += 32) {
    if (k0 + 32 < K) {
      issue(k0 + 32, cur ^ 1);   // prefetch next tile into other buffer
      wait_async_le4();          // <=4 outstanding -> current tile complete
    } else {
      wait_async_0();
    }
    __syncthreads();             // all waves' current-tile data visible

    bf16x16 af[2], bfr[4];
#pragma unroll
    for (int i = 0; i < 2; i++) {
      const unsigned short* base = &lA[cur][(wm * 32 + i * 16 + lr) * 32 + half * 8];
      af[i] = ldfrag(base, base + 16);
    }
#pragma unroll
    for (int j = 0; j < 4; j++) {
      const unsigned short* base = &lB[cur][(wn * 64 + j * 16 + lr) * 32 + half * 8];
      bfr[j] = ldfrag(base, base + 16);
    }
#pragma unroll
    for (int i = 0; i < 2; i++)
#pragma unroll
      for (int j = 0; j < 4; j++)
        acc[i][j] = __builtin_amdgcn_wmma_f32_16x16x32_bf16(
            false, af[i], false, bfr[j], (short)0, acc[i][j], false, false);

    __syncthreads();             // protect buffer being refilled next iteration
    cur ^= 1;
  }

#pragma unroll
  for (int i = 0; i < 2; i++)
#pragma unroll
    for (int j = 0; j < 4; j++) {
      int col = bn * 128 + wn * 64 + j * 16 + lr;
      int rb  = bm * 128 + wm * 32 + i * 16 + half * 8;
#pragma unroll
      for (int r = 0; r < 8; r++) {
        size_t g = (size_t)(rb + r) * N + col;
        float v = acc[i][j][r];
        if (mode == 0) {
          Cf[g] = v;
        } else if (mode == 1) {
          float ge = 0.5f * v * (1.0f + erff(v * 0.70710678118654752f));
          Cbf[g] = f2bf(ge);
        } else {
          Cf[g] = v + resid[g];
        }
      }
    }
}

// LayerNorm (weight-only): fp32 row in, bf16 row out. One block per row.
__global__ __launch_bounds__(256) void k_ln(const float* __restrict__ x,
                                            const float* __restrict__ g,
                                            unsigned short* __restrict__ out)
{
  __shared__ float s1[256], s2[256];
  int row = blockIdx.x, tid = threadIdx.x;
  const float* xr = x + (size_t)row * CC;
  float a0 = xr[tid], a1 = xr[tid + 256];
  s1[tid] = a0 + a1;
  s2[tid] = a0 * a0 + a1 * a1;
  __syncthreads();
  for (int s = 128; s > 0; s >>= 1) {
    if (tid < s) { s1[tid] += s1[tid + s]; s2[tid] += s2[tid + s]; }
    __syncthreads();
  }
  float mu   = s1[0] * (1.0f / CC);
  float var  = s2[0] * (1.0f / CC) - mu * mu;
  float rstd = rsqrtf(var + EPSF);
  out[(size_t)row * CC + tid]       = f2bf((a0 - mu) * rstd * g[tid]);
  out[(size_t)row * CC + tid + 256] = f2bf((a1 - mu) * rstd * g[tid + 256]);
}

__global__ __launch_bounds__(256) void k_embed(const int* __restrict__ idx,
    const float* __restrict__ wte, const float* __restrict__ wpe,
    float* __restrict__ x)
{
  int row = blockIdx.x, tid = threadIdx.x;
  int t = row % TT;
  int tok = idx[row];
  const float* we = wte + (size_t)tok * CC;
  const float* wp = wpe + (size_t)t * CC;
  float* xr = x + (size_t)row * CC;
  xr[tid]       = we[tid] + wp[tid];
  xr[tid + 256] = we[tid + 256] + wp[tid + 256];
}

// Unpack qkv [B*T, 3C] fp32 -> Q[B,H,T,64] (pre-scaled 1/sqrt(D)),
// K[B,H,T,64], V transposed [B,H,64,T], all bf16.
__global__ __launch_bounds__(256) void k_pack(const float* __restrict__ qkv,
    unsigned short* __restrict__ Q, unsigned short* __restrict__ Kp,
    unsigned short* __restrict__ Vt)
{
  int row = blockIdx.x;                // b*T + t
  int b = row / TT, t = row % TT;
  const float* qr = qkv + (size_t)row * 3 * CC;
  for (int c = threadIdx.x; c < CC; c += 256) {
    int h = c >> 6, d = c & 63;
    size_t bh = (size_t)(b * HH + h);
    Q[(bh * TT + t) * 64 + d]  = f2bf(qr[c] * 0.125f);       // 1/sqrt(64)
    Kp[(bh * TT + t) * 64 + d] = f2bf(qr[CC + c]);
    Vt[(bh * 64 + d) * TT + t] = f2bf(qr[2 * CC + c]);
  }
}

__global__ void k_convert(const float* __restrict__ in,
                          unsigned short* __restrict__ out, size_t n)
{
  size_t i = (size_t)blockIdx.x * 256 + threadIdx.x;
  size_t stride = (size_t)gridDim.x * 256;
  for (; i < n; i += stride) out[i] = f2bf(in[i]);
}

// ---------------------------------------------------------------------------
// Top-2 sparse causal attention, flash-style two-pass, one wave per 16-query
// tile. Pass1: S = Q K^T tiles (WMMA), per-row top-2 (over ALL keys, pre-mask,
// matching the reference) + causal max. Pass2: recompute S, mask/exp, LDS
// transpose of P tile into A-fragment layout, P·V via WMMA. y scaled by 1/sum
// afterwards (sum==0 -> 0*inf = NaN, matching reference empty-row softmax).
// ---------------------------------------------------------------------------
__global__ __launch_bounds__(256) void k_attn(const unsigned short* __restrict__ Q,
    const unsigned short* __restrict__ Kp, const unsigned short* __restrict__ Vt,
    unsigned short* __restrict__ ybf)
{
  __shared__ float redA[8][16][16];
  __shared__ float redB[8][16][16];
  __shared__ float redC[8][16][16];
  __shared__ float kthL[8][16], mL[8][16], invL[8][16];
  __shared__ __align__(16) unsigned short ptile[8][16 * 32];

  const int tid = threadIdx.x, lane = tid & 31, wave = tid >> 5;
  const int half = lane >> 4, lr = lane & 15;
  const int bh = blockIdx.y;
  const int b = bh / HH, h = bh % HH;
  const int q0 = (blockIdx.x * 8 + wave) * 16;
  const unsigned short* Qb = Q  + (size_t)bh * TT * 64;
  const unsigned short* Kb = Kp + (size_t)bh * TT * 64;
  const unsigned short* Vb = Vt + (size_t)bh * 64 * TT;

  bf16x16 aq[2];
#pragma unroll
  for (int kk = 0; kk < 2; kk++) {
    const unsigned short* base = Qb + (size_t)(q0 + lr) * 64 + kk * 32 + half * 8;
    aq[kk] = ldfrag(base, base + 16);
  }

  // ---- pass 1: per-row top-2 over all keys + causal max ----
  float m1[8], m2[8], cm[8];
#pragma unroll
  for (int r = 0; r < 8; r++) { m1[r] = NEG_INF; m2[r] = NEG_INF; cm[r] = NEG_INF; }

  for (int s0 = 0; s0 < TT; s0 += 16) {
    f32x8 sacc = {};
#pragma unroll
    for (int kk = 0; kk < 2; kk++) {
      const unsigned short* base = Kb + (size_t)(s0 + lr) * 64 + kk * 32 + half * 8;
      bf16x16 bk = ldfrag(base, base + 16);
      sacc = __builtin_amdgcn_wmma_f32_16x16x32_bf16(
          false, aq[kk], false, bk, (short)0, sacc, false, false);
    }
    int scol = s0 + lr;
#pragma unroll
    for (int r = 0; r < 8; r++) {
      float v = sacc[r];
      if (v > m1[r]) { m2[r] = m1[r]; m1[r] = v; }
      else if (v > m2[r]) m2[r] = v;
      int qrow = q0 + r + 8 * half;
      if (scol <= qrow) cm[r] = fmaxf(cm[r], v);
    }
  }
#pragma unroll
  for (int r = 0; r < 8; r++) {
    int rr = r + 8 * half;
    redA[wave][rr][lr] = m1[r];
    redB[wave][rr][lr] = m2[r];
    redC[wave][rr][lr] = cm[r];
  }
  __syncthreads();
  if (lane < 16) {
    float a1 = NEG_INF, a2 = NEG_INF, cmx = NEG_INF;
#pragma unroll
    for (int c = 0; c < 16; c++) {
      float b1 = redA[wave][lane][c], b2 = redB[wave][lane][c];
      float n1 = fmaxf(a1, b1);
      float n2 = fmaxf(fminf(a1, b1), fmaxf(a2, b2));
      a1 = n1; a2 = n2;
      cmx = fmaxf(cmx, redC[wave][lane][c]);
    }
    kthL[wave][lane] = a2;                       // 2nd-largest = top-k threshold
    mL[wave][lane]   = (cmx >= a2) ? cmx : 0.0f; // surviving max (unused if none)
  }
  __syncthreads();
  float kth_r[8], m_r[8];
#pragma unroll
  for (int r = 0; r < 8; r++) {
    int rr = r + 8 * half;
    kth_r[r] = kthL[wave][rr];
    m_r[r]   = mL[wave][rr];
  }

  // ---- pass 2: recompute S, mask+exp, P·V via WMMA ----
  f32x8 oacc[4] = {};
  float rsum[8] = {};
  for (int s0 = 0; s0 < TT; s0 += 32) {
#pragma unroll
    for (int t = 0; t < 2; t++) {
      f32x8 sacc = {};
#pragma unroll
      for (int kk = 0; kk < 2; kk++) {
        const unsigned short* base =
            Kb + (size_t)(s0 + t * 16 + lr) * 64 + kk * 32 + half * 8;
        bf16x16 bk = ldfrag(base, base + 16);
        sacc = __builtin_amdgcn_wmma_f32_16x16x32_bf16(
            false, aq[kk], false, bk, (short)0, sacc, false, false);
      }
      int scol = s0 + t * 16 + lr;
#pragma unroll
      for (int r = 0; r < 8; r++) {
        int qrow = q0 + r + 8 * half;
        float v = sacc[r];
        float e = ((v >= kth_r[r]) && (scol <= qrow)) ? __expf(v - m_r[r]) : 0.0f;
        rsum[r] += e;
        ptile[wave][(r + 8 * half) * 32 + t * 16 + lr] = f2bf(e);
      }
    }
    __syncthreads();
    const unsigned short* pbase = &ptile[wave][lr * 32 + half * 8];
    bf16x16 pa = ldfrag(pbase, pbase + 16);
#pragma unroll
    for (int j = 0; j < 4; j++) {
      const unsigned short* vb = Vb + (size_t)(j * 16 + lr) * TT + s0 + half * 8;
      bf16x16 bv = ldfrag(vb, vb + 16);
      oacc[j] = __builtin_amdgcn_wmma_f32_16x16x32_bf16(
          false, pa, false, bv, (short)0, oacc[j], false, false);
    }
    __syncthreads();
  }

  // row sums -> 1/sum
#pragma unroll
  for (int r = 0; r < 8; r++) redA[wave][r + 8 * half][lr] = rsum[r];
  __syncthreads();
  if (lane < 16) {
    float s = 0.f;
#pragma unroll
    for (int c = 0; c < 16; c++) s += redA[wave][lane][c];
    invL[wave][lane] = 1.0f / s;                 // inf if empty row -> NaN y
  }
  __syncthreads();
#pragma unroll
  for (int j = 0; j < 4; j++) {
    int d = j * 16 + lr;
#pragma unroll
    for (int r = 0; r < 8; r++) {
      int qrow = q0 + r + 8 * half;
      float y = oacc[j][r] * invL[wave][r + 8 * half];
      size_t rowg = (size_t)(b * TT + qrow);
      ybf[rowg * CC + h * 64 + d] = f2bf(y);
    }
  }
}

// ---------------------------------------------------------------------------
extern "C" void kernel_launch(void* const* d_in, const int* in_sizes, int n_in,
                              void* d_out, int out_size, void* d_ws, size_t ws_size,
                              hipStream_t stream)
{
  const int*   idx   = (const int*)d_in[0];
  const float* wte   = (const float*)d_in[1];
  const float* wpe   = (const float*)d_in[2];
  const float* ln1g  = (const float*)d_in[3];
  const float* attnw = (const float*)d_in[4];
  const float* projw = (const float*)d_in[5];
  const float* ln2g  = (const float*)d_in[6];
  const float* fcw   = (const float*)d_in[7];
  const float* mlpw  = (const float*)d_in[8];
  const float* lnfg  = (const float*)d_in[9];
  float* out = (float*)d_out;

  char* ws = (char*)d_ws;
  size_t off = 0;
  auto take = [&](size_t bytes) -> char* {
    char* p = ws + off;
    off = (off + bytes + 255) & ~(size_t)255;
    return p;
  };
  float*          x     = (float*)take((size_t)BT * CC * 4);
  unsigned short* xnbf  = (unsigned short*)take((size_t)BT * CC * 2);
  float*          qkvf  = (float*)take((size_t)BT * 3 * CC * 4);
  unsigned short* Qb    = (unsigned short*)take((size_t)BT * CC * 2);
  unsigned short* Kb    = (unsigned short*)take((size_t)BT * CC * 2);
  unsigned short* Vt    = (unsigned short*)take((size_t)BT * CC * 2);
  unsigned short* ybf   = (unsigned short*)take((size_t)BT * CC * 2);
  unsigned short* hbf   = (unsigned short*)take((size_t)BT * 4 * CC * 2);
  unsigned short* wattn = (unsigned short*)take((size_t)LNUM * 3 * CC * CC * 2);
  unsigned short* wproj = (unsigned short*)take((size_t)LNUM * CC * CC * 2);
  unsigned short* wfc   = (unsigned short*)take((size_t)LNUM * 4 * CC * CC * 2);
  unsigned short* wmlp  = (unsigned short*)take((size_t)LNUM * CC * 4 * CC * 2);
  unsigned short* wtebf = (unsigned short*)take((size_t)VV * CC * 2);

  auto conv = [&](const float* src, unsigned short* dst, size_t n) {
    int grid = (int)((n + 255) / 256);
    if (grid > 16384) grid = 16384;
    k_convert<<<grid, 256, 0, stream>>>(src, dst, n);
  };
  conv(attnw, wattn, (size_t)LNUM * 3 * CC * CC);
  conv(projw, wproj, (size_t)LNUM * CC * CC);
  conv(fcw,   wfc,   (size_t)LNUM * 4 * CC * CC);
  conv(mlpw,  wmlp,  (size_t)LNUM * CC * 4 * CC);
  conv(wte,   wtebf, (size_t)VV * CC);

  k_embed<<<BT, 256, 0, stream>>>(idx, wte, wpe, x);

  for (int l = 0; l < LNUM; l++) {
    k_ln<<<BT, 256, 0, stream>>>(x, ln1g + (size_t)l * CC, xnbf);
    k_gemm<<<dim3(3 * CC / 128, BT / 128), 256, 0, stream>>>(
        xnbf, wattn + (size_t)l * 3 * CC * CC, qkvf, nullptr, nullptr,
        BT, 3 * CC, CC, 0);
    k_pack<<<BT, 256, 0, stream>>>(qkvf, Qb, Kb, Vt);
    k_attn<<<dim3(TT / 128, BB * HH), 256, 0, stream>>>(Qb, Kb, Vt, ybf);
    k_gemm<<<dim3(CC / 128, BT / 128), 256, 0, stream>>>(
        ybf, wproj + (size_t)l * CC * CC, x, nullptr, x, BT, CC, CC, 2);
    k_ln<<<BT, 256, 0, stream>>>(x, ln2g + (size_t)l * CC, xnbf);
    k_gemm<<<dim3(4 * CC / 128, BT / 128), 256, 0, stream>>>(
        xnbf, wfc + (size_t)l * 4 * CC * CC, nullptr, hbf, nullptr,
        BT, 4 * CC, CC, 1);
    k_gemm<<<dim3(CC / 128, BT / 128), 256, 0, stream>>>(
        hbf, wmlp + (size_t)l * CC * 4 * CC, x, nullptr, x, BT, CC, 4 * CC, 2);
  }
  k_ln<<<BT, 256, 0, stream>>>(x, lnfg, xnbf);
  k_gemm<<<dim3(VV / 128, BT / 128), 256, 0, stream>>>(
      xnbf, wtebf, out, nullptr, nullptr, BT, VV, CC, 0);

  (void)in_sizes; (void)n_in; (void)out_size; (void)ws_size;
}
